// AffineField4D_81046032875686
// MI455X (gfx1250) — compile-verified, matrix operand order checked by reading, not run on hardware
//
#include <hip/hip_runtime.h>

typedef __attribute__((ext_vector_type(16))) _Float16 v16h;
typedef __attribute__((ext_vector_type(8)))  float    v8f;

#define HD      256
#define SWS     (HD + 16)   // padded LDS row stride (halves) -> no bank conflicts
#define PTS     32          // points per block
#define ROWS    128         // 4 streams (value + 3 tangents) * PTS
#define NW      8           // waves per block (wave32)
#define OMEGA0  30.0f
#define GMC     0.03f
#define NPTS    100000
#define NBLOCKS 3125        // NPTS / PTS exactly

// d_ws layout: f16 weights [W1h 65536][W2h 65536][W3h 65536][W4h 4096 padded], then f32 partials[NBLOCKS]
#define WS_HALVES (3*65536 + 4096)

// ---------------------------------------------------------------- convert ----
__global__ void convert_weights_kernel(const float* __restrict__ W1,
                                       const float* __restrict__ W2,
                                       const float* __restrict__ W3,
                                       const float* __restrict__ W4,
                                       _Float16* __restrict__ out) {
  int i = blockIdx.x * 256 + threadIdx.x;
  if (i >= WS_HALVES) return;
  if (i < 65536)        out[i] = (_Float16)W1[i];
  else if (i < 131072)  out[i] = (_Float16)W2[i - 65536];
  else if (i < 196608)  out[i] = (_Float16)W3[i - 131072];
  else {
    int j = i - 196608;                 // padded 16x256, rows 12..15 are zero
    out[i] = ((j >> 8) < 12) ? (_Float16)W4[j] : (_Float16)0.0f;
  }
}

// ------------------------------------------------------------------- main ----
__global__ __launch_bounds__(256) void siren_deform_kernel(
    const float* __restrict__ xyz, const float* __restrict__ ts,
    const int* __restrict__ tidx,
    const float* __restrict__ W0, const float* __restrict__ b0,
    const float* __restrict__ b1, const float* __restrict__ b2,
    const float* __restrict__ b3, const float* __restrict__ b4,
    const _Float16* __restrict__ Wh,
    float* __restrict__ pred,           // d_out[0 .. 3N)
    float* __restrict__ partial) {      // per-block loss partials in d_ws
  __shared__ __align__(32) _Float16 sA[ROWS][SWS];   // ~68 KB activation tile (padded)
  __shared__ __align__(32) _Float16 sW[HD * SWS];    // ~136 KB weight stage (padded)
  __shared__ float sFrag[NW][256];                   // 8 KB per-wave f32 frag scratch
  __shared__ float sXYZ[PTS][3];
  __shared__ float sContrib[PTS];

  const int tid    = threadIdx.x;
  const int wave   = tid >> 5;
  const int lane   = tid & 31;
  const int lane15 = lane & 15;
  const int row0   = wave * 16;
  const int p0     = blockIdx.x * PTS;

  if (tid < PTS * 3) sXYZ[tid / 3][tid % 3] = xyz[p0 * 3 + tid];
  const float tval = ts[tidx[0]];
  __syncthreads();

  // ---- layer 0 (K=4, scalar): value + 3 tangent streams, interleaved rows 4p+s
  {
    const int o = tid;
    const float w0 = W0[o * 4 + 0], w1 = W0[o * 4 + 1];
    const float w2 = W0[o * 4 + 2], w3 = W0[o * 4 + 3];
    const float bb = b0[o];
    for (int p = 0; p < PTS; ++p) {
      float z = w0 * sXYZ[p][0] + w1 * sXYZ[p][1] + w2 * sXYZ[p][2] + w3 * tval + bb;
      float s = __sinf(OMEGA0 * z);
      float c = OMEGA0 * __cosf(OMEGA0 * z);
      sA[4 * p + 0][o] = (_Float16)s;
      sA[4 * p + 1][o] = (_Float16)(c * w0);
      sA[4 * p + 2][o] = (_Float16)(c * w1);
      sA[4 * p + 3][o] = (_Float16)(c * w2);
    }
  }
  __syncthreads();

  const v8f vzero = {0.f, 0.f, 0.f, 0.f, 0.f, 0.f, 0.f, 0.f};
  const int arow  = row0 + lane15;
  const int khalf = (lane >> 4) * 16;
  const float* biases[3] = {b1, b2, b3};

  // ---- hidden layers 1..3: 128x256x256 GEMM per block via WMMA f16
#pragma unroll 1
  for (int L = 0; L < 3; ++L) {
    {  // stage 256x256 f16 weights -> padded LDS rows (row-major [out][in])
      const uint4* s4 = (const uint4*)(Wh + L * 65536);
      for (int i = tid; i < 8192; i += 256) {
        int r = i >> 5, c = i & 31;
        *((uint4*)&sW[r * SWS] + c) = s4[i];
      }
    }
    // prefetch next layer's weights into cache while this layer computes
    {
      const char* nxt = (const char*)(Wh + (L + 1) * 65536);
      const int span = (L < 2) ? 131072 : 8192;        // bytes
      for (int off = tid * 256; off < span; off += 256 * 256)
        __builtin_prefetch(nxt + off, 0, 1);
    }
    __syncthreads();

    v8f acc[16];
#pragma unroll
    for (int n = 0; n < 16; ++n) acc[n] = vzero;

#pragma unroll
    for (int k = 0; k < 8; ++k) {
      v16h a = *(const v16h*)&sA[arow][k * 32 + khalf];
      v16h bf[16];
#pragma unroll
      for (int n = 0; n < 16; ++n)       // issue all 16 B-frag loads first
        bf[n] = *(const v16h*)&sW[(n * 16 + lane15) * SWS + k * 32 + khalf];
#pragma unroll
      for (int n = 0; n < 16; ++n)       // then back-to-back WMMAs
        acc[n] = __builtin_amdgcn_wmma_f32_16x16x32_f16(
            false, a, false, bf[n], (short)0, acc[n], false, false);
    }

    // elementwise: value rows -> sin(w*(z+b)); tangent rows -> w*cos(w*(zv+b))*z
    const float* bL = biases[L];
#pragma unroll 1
    for (int n = 0; n < 16; ++n) {
#pragma unroll
      for (int r = 0; r < 8; ++r) {
        int row = r + ((lane >> 4) << 3);
        sFrag[wave][row * 16 + lane15] = acc[n][r];
      }
      asm volatile("s_wait_dscnt 0" ::: "memory");
#pragma unroll
      for (int i = 0; i < 8; ++i) {
        int idx = i * 32 + lane;
        int r = idx >> 4, c = idx & 15;
        int o = n * 16 + c;
        float z    = sFrag[wave][r * 16 + c];
        float zv   = sFrag[wave][(r & ~3) * 16 + c] + bL[o];
        float outv = ((r & 3) == 0) ? __sinf(OMEGA0 * zv)
                                    : OMEGA0 * __cosf(OMEGA0 * zv) * z;
        sA[row0 + r][o] = (_Float16)outv;
      }
      asm volatile("s_wait_dscnt 0" ::: "memory");
    }
    __syncthreads();
  }

  // ---- final layer: 128x256 @ 256x16 (12 valid cols), one WMMA col-tile
  {
    const uint4* s4 = (const uint4*)(Wh + 3 * 65536);
    for (int i = tid; i < 512; i += 256) {
      int r = i >> 5, c = i & 31;
      *((uint4*)&sW[r * SWS] + c) = s4[i];
    }
  }
  __syncthreads();

  v8f acc4 = vzero;
#pragma unroll
  for (int k = 0; k < 8; ++k) {
    v16h a = *(const v16h*)&sA[arow][k * 32 + khalf];
    v16h b = *(const v16h*)&sW[lane15 * SWS + k * 32 + khalf];
    acc4 = __builtin_amdgcn_wmma_f32_16x16x32_f16(
        false, a, false, b, (short)0, acc4, false, false);
  }
#pragma unroll
  for (int r = 0; r < 8; ++r) {
    int row = r + ((lane >> 4) << 3);
    sFrag[wave][row * 16 + lane15] = acc4[r];
  }
  asm volatile("s_wait_dscnt 0" ::: "memory");

  if (lane < 4) {
    const int pl = wave * 4 + lane;       // point in block
    const int pg = p0 + pl;
    const float x0 = sXYZ[pl][0], x1 = sXYZ[pl][1], x2 = sXYZ[pl][2];
    float outv[12], outt[3][12];
#pragma unroll
    for (int j = 0; j < 12; ++j)
      outv[j] = sFrag[wave][(4 * lane + 0) * 16 + j] + b4[j];
#pragma unroll
    for (int s = 1; s < 4; ++s)
#pragma unroll
      for (int j = 0; j < 12; ++j)
        outt[s - 1][j] = sFrag[wave][(4 * lane + s) * 16 + j];

    // xyz_pred = A x + u
#pragma unroll
    for (int i = 0; i < 3; ++i)
      pred[pg * 3 + i] = outv[3 * i + 0] * x0 + outv[3 * i + 1] * x1 +
                         outv[3 * i + 2] * x2 + outv[9 + i];

    // J_ik = A_ik + sum_j x_j dA_ij/dx_k + du_i/dx_k
    float J[3][3];
#pragma unroll
    for (int i = 0; i < 3; ++i)
#pragma unroll
      for (int k = 0; k < 3; ++k)
        J[i][k] = outv[3 * i + k] + x0 * outt[k][3 * i + 0] +
                  x1 * outt[k][3 * i + 1] + x2 * outt[k][3 * i + 2] +
                  outt[k][9 + i];

    // M = J^T J (symmetric), eigenvalues via trigonometric closed form
    float m00 = J[0][0]*J[0][0] + J[1][0]*J[1][0] + J[2][0]*J[2][0];
    float m11 = J[0][1]*J[0][1] + J[1][1]*J[1][1] + J[2][1]*J[2][1];
    float m22 = J[0][2]*J[0][2] + J[1][2]*J[1][2] + J[2][2]*J[2][2];
    float m01 = J[0][0]*J[0][1] + J[1][0]*J[1][1] + J[2][0]*J[2][1];
    float m02 = J[0][0]*J[0][2] + J[1][0]*J[1][2] + J[2][0]*J[2][2];
    float m12 = J[0][1]*J[0][2] + J[1][1]*J[1][2] + J[2][1]*J[2][2];

    float p1 = m01*m01 + m02*m02 + m12*m12;
    float q  = (m00 + m11 + m22) * (1.0f / 3.0f);
    float d0 = m00 - q, d1 = m11 - q, d2v = m22 - q;
    float p2 = d0*d0 + d1*d1 + d2v*d2v + 2.0f * p1;
    float p  = __fsqrt_rn(p2 * (1.0f / 6.0f)) + 1e-20f;
    float ip = 1.0f / p;
    float b00 = d0*ip, b11 = d1*ip, b22 = d2v*ip;
    float b01 = m01*ip, b02 = m02*ip, b12 = m12*ip;
    float detB = b00*(b11*b22 - b12*b12) - b01*(b01*b22 - b12*b02)
               + b02*(b01*b12 - b11*b02);
    float rr  = fminf(fmaxf(0.5f * detB, -1.0f), 1.0f);
    float phi = acosf(rr) * (1.0f / 3.0f);
    float e1 = q + 2.0f * p * __cosf(phi);
    float e3 = q + 2.0f * p * __cosf(phi + 2.0943951023931953f);
    float e2 = 3.0f * q - e1 - e3;

    float l1 = 0.5f * __logf(fmaxf(e1, 1e-20f));
    float l2 = 0.5f * __logf(fmaxf(e2, 1e-20f));
    float l3 = 0.5f * __logf(fmaxf(e3, 1e-20f));
    float rsum = l1*l1 + l2*l2 + l3*l3;
    float dd = (rsum * (1.0f / GMC)) * (rsum * (1.0f / GMC));
    sContrib[pl] = 2.0f * dd / (dd + 4.0f);
  }
  __syncthreads();
  if (tid == 0) {
    float s = 0.0f;
    for (int pI = 0; pI < PTS; ++pI) s += sContrib[pI];   // fixed order: deterministic
    partial[blockIdx.x] = s;
  }
}

// ----------------------------------------------------------------- reduce ----
__global__ void reduce_loss_kernel(const float* __restrict__ partial,
                                   float* __restrict__ out_loss) {
  __shared__ float sm[256];
  float s = 0.0f;
  for (int i = threadIdx.x; i < NBLOCKS; i += 256) s += partial[i];
  sm[threadIdx.x] = s;
  __syncthreads();
  for (int off = 128; off > 0; off >>= 1) {
    if (threadIdx.x < off) sm[threadIdx.x] += sm[threadIdx.x + off];
    __syncthreads();
  }
  if (threadIdx.x == 0) out_loss[0] = sm[0] * (1.0f / (float)NPTS);
}

// ----------------------------------------------------------------- launch ----
extern "C" void kernel_launch(void* const* d_in, const int* in_sizes, int n_in,
                              void* d_out, int out_size, void* d_ws, size_t ws_size,
                              hipStream_t stream) {
  const float* xyz = (const float*)d_in[0];
  const float* ts  = (const float*)d_in[1];
  const int*   tix = (const int*)d_in[2];
  const float* W0  = (const float*)d_in[3];
  const float* b0  = (const float*)d_in[4];
  const float* W1  = (const float*)d_in[5];
  const float* b1  = (const float*)d_in[6];
  const float* W2  = (const float*)d_in[7];
  const float* b2  = (const float*)d_in[8];
  const float* W3  = (const float*)d_in[9];
  const float* b3  = (const float*)d_in[10];
  const float* W4  = (const float*)d_in[11];
  const float* b4  = (const float*)d_in[12];

  _Float16* Wh     = (_Float16*)d_ws;
  float*    parts  = (float*)((char*)d_ws + WS_HALVES * sizeof(_Float16));
  float*    out    = (float*)d_out;

  convert_weights_kernel<<<(WS_HALVES + 255) / 256, 256, 0, stream>>>(W1, W2, W3, W4, Wh);
  siren_deform_kernel<<<NBLOCKS, 256, 0, stream>>>(xyz, ts, tix, W0, b0, b1, b2, b3, b4,
                                                   Wh, out, parts);
  reduce_loss_kernel<<<1, 256, 0, stream>>>(parts, out + 3 * NPTS);
}
